// SelfAttentionBlock_52158082842846
// MI455X (gfx1250) — matrix operand order repaired and checked by reading, not compiled, not run on hardware
//
#include <hip/hip_runtime.h>

// ---------------------------------------------------------------------------
// Types / helpers for CDNA5 (gfx1250) bf16 WMMA
// ---------------------------------------------------------------------------
typedef unsigned short u16;
typedef __attribute__((ext_vector_type(8)))  unsigned short v8u;
typedef __attribute__((ext_vector_type(16))) unsigned short v16u;
typedef __attribute__((ext_vector_type(16))) __bf16         v16bf;
typedef __attribute__((ext_vector_type(8)))  float          v8f;
typedef __attribute__((ext_vector_type(4)))  float          v4f;
typedef __attribute__((ext_vector_type(4)))  int            v4i;

// Native bf16 convert (RNE); backend picks v_cvt_pk_bf16_f32-class ops.
__device__ __forceinline__ u16 f2bf(float f) {
    return __builtin_bit_cast(u16, static_cast<__bf16>(f));
}

__device__ __forceinline__ v8f vzero8() {
    v8f z;
#pragma unroll
    for (int i = 0; i < 8; ++i) z[i] = 0.0f;
    return z;
}

__device__ __forceinline__ v8u pack8(const u16* t) {
    v8u v;
#pragma unroll
    for (int i = 0; i < 8; ++i) v[i] = t[i];
    return v;
}

// Assemble a 16-element bf16 fragment from two 8-element (b128) loads.
__device__ __forceinline__ v16bf mkfrag(v8u lo, v8u hi) {
    v16u t;
#pragma unroll
    for (int i = 0; i < 8; ++i) { t[i] = lo[i]; t[i + 8] = hi[i]; }
    return __builtin_bit_cast(v16bf, t);
}

__device__ __forceinline__ v8f wmma_bf16(v16bf a, v16bf b, v8f c) {
    // D(f32 16x16) = A(bf16 16x32) * B(bf16 32x16) + C
    return __builtin_amdgcn_wmma_f32_16x16x32_bf16(
        /*neg_a=*/false, a, /*neg_b=*/false, b,
        /*c_mod=*/(short)0, c, /*reuse_a=*/false, /*reuse_b=*/false);
}

// ---------------------------------------------------------------------------
// Async global->LDS copy (GLOBAL_LOAD_ASYNC_TO_LDS_B128, ASYNCcnt-tracked).
// Signature (from hipcc diagnostic): (int4 as1*, int4 as3*, imm offset, imm cpol)
// Guarded: falls back to synchronous VGPR staging if the builtin is absent.
// ---------------------------------------------------------------------------
#if __has_builtin(__builtin_amdgcn_global_load_async_to_lds_b128)
#define HAS_ASYNC_LDS 1
typedef __attribute__((address_space(1))) v4i* gv4i_p;
typedef __attribute__((address_space(3))) v4i* lv4i_p;
__device__ __forceinline__ void async_copy16(const void* g, void* l) {
    __builtin_amdgcn_global_load_async_to_lds_b128(
        (gv4i_p)g, (lv4i_p)l, /*offset=*/0, /*cpol=*/0);
}
#define ASYNC_WAIT() asm volatile("s_wait_asynccnt 0" ::: "memory")
#else
#define HAS_ASYNC_LDS 0
#define ASYNC_WAIT()
#endif

// ---------------------------------------------------------------------------
// Tiled GEMM:  C[M,N] = A[M,K] @ B[K,N] + bias[N]
//   AT = float (fp32 input, converted to bf16 in staging) or u16 (bf16 input)
//   OT = u16 (bf16 output) or float (fp32 output)
// Block tile 128x128, K-slab 32, 256 threads = 8 wave32 waves (2x4 grid),
// each wave computes a 64x32 region = eight 16x16 WMMA accumulators.
// ---------------------------------------------------------------------------
static constexpr int GBM = 128, GBN = 128, GBK = 32, GLD = 40; // GLD*2B = 80B (16B aligned rows)

template <typename AT, typename OT>
__global__ __launch_bounds__(256) void gemm_wmma(
    const AT* __restrict__ A, const float* __restrict__ B,
    const float* __restrict__ bias, OT* __restrict__ C,
    int M, int N, int K)
{
    __shared__ u16 Al[GBM * GLD];   // [row][k]  row-major,  padded
    __shared__ u16 Bl[GBN * GLD];   // [col][k]  B transposed, padded

    const int tid  = threadIdx.x;
    const int lane = tid & 31, w = tid >> 5;
    const int half = lane >> 4, l16 = lane & 15;
    const int wm = w >> 2, wn = w & 3;           // 2 x 4 wave grid
    const int m0 = blockIdx.y * GBM, n0 = blockIdx.x * GBN;

    v8f acc[4][2];
#pragma unroll
    for (int mi = 0; mi < 4; ++mi)
#pragma unroll
        for (int ni = 0; ni < 2; ++ni) acc[mi][ni] = vzero8();

    for (int kb = 0; kb < K; kb += GBK) {
        // ---- stage A (128 x 32) as bf16 ----
        {
            const int row = tid >> 1, c0 = (tid & 1) * 16;
            const AT* src = A + (size_t)(m0 + row) * K + kb + c0;
            if constexpr (sizeof(AT) == 4) {
                u16 tmp[16];
#pragma unroll
                for (int i = 0; i < 4; ++i) {
                    v4f v = *(const v4f*)((const float*)src + 4 * i);
                    tmp[4*i+0] = f2bf(v[0]); tmp[4*i+1] = f2bf(v[1]);
                    tmp[4*i+2] = f2bf(v[2]); tmp[4*i+3] = f2bf(v[3]);
                }
                *(v8u*)&Al[row * GLD + c0]     = pack8(tmp);
                *(v8u*)&Al[row * GLD + c0 + 8] = pack8(tmp + 8);
            } else {
#if HAS_ASYNC_LDS
                // bf16 input: straight byte copy -> async DMA into LDS
                async_copy16((const u16*)src,     &Al[row * GLD + c0]);
                async_copy16((const u16*)src + 8, &Al[row * GLD + c0 + 8]);
#else
                v8u a0 = *(const v8u*)((const u16*)src);
                v8u a1 = *(const v8u*)((const u16*)src + 8);
                *(v8u*)&Al[row * GLD + c0]     = a0;
                *(v8u*)&Al[row * GLD + c0 + 8] = a1;
#endif
            }
        }
        // ---- stage B (32 x 128), convert + transpose -> Bl[col][k] ----
        {
            const int kr = tid >> 3, c0 = (tid & 7) * 16;
            const float* src = B + (size_t)(kb + kr) * N + n0 + c0;
#pragma unroll
            for (int i = 0; i < 16; i += 4) {
                v4f v = *(const v4f*)(src + i);
                Bl[(c0 + i + 0) * GLD + kr] = f2bf(v[0]);
                Bl[(c0 + i + 1) * GLD + kr] = f2bf(v[1]);
                Bl[(c0 + i + 2) * GLD + kr] = f2bf(v[2]);
                Bl[(c0 + i + 3) * GLD + kr] = f2bf(v[3]);
            }
        }
#if HAS_ASYNC_LDS
        if constexpr (sizeof(AT) == 2) { ASYNC_WAIT(); }
#endif
        __syncthreads();

        // ---- fragments + 8 WMMAs ----
        v16bf bfrag[2];
#pragma unroll
        for (int ni = 0; ni < 2; ++ni) {
            // B layout: lanes 0-15 -> K 0..15, lanes 16-31 -> K 16..31
            const u16* p = &Bl[(wn * 32 + ni * 16 + l16) * GLD + half * 16];
            bfrag[ni] = mkfrag(*(const v8u*)p, *(const v8u*)(p + 8));
        }
#pragma unroll
        for (int mi = 0; mi < 4; ++mi) {
            // A layout: lanes 0-15 -> K 0..15, lanes 16-31 -> K 16..31
            const u16* p = &Al[(wm * 64 + mi * 16 + l16) * GLD + half * 16];
            v16bf afrag = mkfrag(*(const v8u*)p, *(const v8u*)(p + 8));
            acc[mi][0] = wmma_bf16(afrag, bfrag[0], acc[mi][0]);
            acc[mi][1] = wmma_bf16(afrag, bfrag[1], acc[mi][1]);
        }
        __syncthreads();
    }

    // ---- epilogue: bias + store (C layout: VGPR r -> row r / r+8, lane -> col) ----
#pragma unroll
    for (int ni = 0; ni < 2; ++ni) {
        const int col = n0 + wn * 32 + ni * 16 + l16;
        const float bv = bias[col];
#pragma unroll
        for (int mi = 0; mi < 4; ++mi) {
#pragma unroll
            for (int r = 0; r < 8; ++r) {
                const int row = m0 + wm * 64 + mi * 16 + r + half * 8;
                const float v = acc[mi][ni][r] + bv;
                if constexpr (sizeof(OT) == 2) C[(size_t)row * N + col] = f2bf(v);
                else                           C[(size_t)row * N + col] = v;
            }
        }
    }
}

// ---------------------------------------------------------------------------
// Flash attention over bf16 projections.
//   Qp/Kp/Vp: [B*S, 2048] bf16, head h occupies cols h*128 .. h*128+127.
//   grid = (B*H, S/128); block = 256 (8 waves); each wave owns 16 query rows.
// ---------------------------------------------------------------------------
static constexpr int A_S = 2048, A_D = 2048, A_DK = 128, A_KT = 32;
static constexpr int LDK = 136;   // 272B rows (16B aligned)
static constexpr int LDV = 40;    // 80B rows  (16B aligned)

__global__ __launch_bounds__(256) void attn_wmma(
    const u16* __restrict__ Qp, const u16* __restrict__ Kp,
    const u16* __restrict__ Vp, u16* __restrict__ O)
{
    __shared__ u16 Kl[A_KT * LDK];      // [key][d]
    __shared__ u16 Vt[A_DK * LDV];      // [d][key]  (transposed)
    __shared__ u16 Pl[8 * 16 * 32];     // per-wave P scratch (16x32)

    const int tid = threadIdx.x, lane = tid & 31, w = tid >> 5;
    const int half = lane >> 4, l16 = lane & 15;
    const int bh = blockIdx.x, b = bh >> 4, h = bh & 15;
    const int q0 = blockIdx.y * 128 + w * 16;
    const float scale = 0.08838834764831845f;   // 1/sqrt(128)

    const size_t baserow = (size_t)b * A_S;
    const size_t headoff = (size_t)h * A_DK;

    // Q fragments (per-wave, 16 rows x 128 d), loaded straight in A-layout.
    v16bf qf[4];
    {
        const u16* qrow = Qp + (baserow + q0 + l16) * (size_t)A_D + headoff;
#pragma unroll
        for (int c = 0; c < 4; ++c) {
            const u16* p = qrow + 32 * c + 16 * half;
            qf[c] = mkfrag(*(const v8u*)p, *(const v8u*)(p + 8));
        }
    }

    v8f o[8];
    float m[8], l[8];
#pragma unroll
    for (int j = 0; j < 8; ++j) o[j] = vzero8();
#pragma unroll
    for (int r = 0; r < 8; ++r) { m[r] = -1e30f; l[r] = 0.0f; }

    for (int kt = 0; kt < A_S / A_KT; ++kt) {
        const int j0 = kt * A_KT;
        // ---- cooperative stage: K tile via async DMA, V tile transposed ----
        {
            const int row = tid >> 3, c0 = (tid & 7) * 16;   // 32 keys x 128 d
            const u16* ks = Kp + (baserow + j0 + row) * (size_t)A_D + headoff + c0;
#if HAS_ASYNC_LDS
            async_copy16(ks,     &Kl[row * LDK + c0]);
            async_copy16(ks + 8, &Kl[row * LDK + c0 + 8]);
#else
            v8u ka = *(const v8u*)ks, kb2 = *(const v8u*)(ks + 8);
            *(v8u*)&Kl[row * LDK + c0]     = ka;
            *(v8u*)&Kl[row * LDK + c0 + 8] = kb2;
#endif
            const u16* vs = Vp + (baserow + j0 + row) * (size_t)A_D + headoff + c0;
            v8u va = *(const v8u*)vs, vb = *(const v8u*)(vs + 8);
#pragma unroll
            for (int i = 0; i < 8; ++i) {
                Vt[(c0 + i)     * LDV + row] = va[i];
                Vt[(c0 + 8 + i) * LDV + row] = vb[i];
            }
        }
        ASYNC_WAIT();
        __syncthreads();

        // ---- S = Q @ K^T (two 16x16 tiles, K-dim 128 in 4 WMMAs each) ----
        v8f s0 = vzero8(), s1 = vzero8();
#pragma unroll
        for (int c = 0; c < 4; ++c) {
            const u16* p0 = &Kl[l16 * LDK + 32 * c + 16 * half];
            s0 = wmma_bf16(qf[c], mkfrag(*(const v8u*)p0, *(const v8u*)(p0 + 8)), s0);
            const u16* p1 = &Kl[(16 + l16) * LDK + 32 * c + 16 * half];
            s1 = wmma_bf16(qf[c], mkfrag(*(const v8u*)p1, *(const v8u*)(p1 + 8)), s1);
        }

        // ---- online softmax (row r lives in lanes 0-15 / r+8 in 16-31) ----
        float alpha[8];
#pragma unroll
        for (int r = 0; r < 8; ++r) {
            const float a0 = s0[r] * scale, a1 = s1[r] * scale;
            float t = fmaxf(a0, a1);
#pragma unroll
            for (int mm = 1; mm < 16; mm <<= 1) t = fmaxf(t, __shfl_xor(t, mm, 32));
            const float mn = fmaxf(m[r], t);
            const float al = __expf(m[r] - mn);
            const float p0v = __expf(a0 - mn), p1v = __expf(a1 - mn);
            float rs = p0v + p1v;
#pragma unroll
            for (int mm = 1; mm < 16; mm <<= 1) rs += __shfl_xor(rs, mm, 32);
            m[r] = mn; l[r] = l[r] * al + rs; alpha[r] = al;
            s0[r] = p0v; s1[r] = p1v;
        }
#pragma unroll
        for (int j = 0; j < 8; ++j)
#pragma unroll
            for (int r = 0; r < 8; ++r) o[j][r] *= alpha[r];

        // ---- re-layout P (C layout -> A layout) via per-wave LDS scratch ----
        u16* pw = &Pl[w * 16 * 32];
#pragma unroll
        for (int r = 0; r < 8; ++r) {
            const int row = r + 8 * half;
            pw[row * 32 + l16]      = f2bf(s0[r]);
            pw[row * 32 + 16 + l16] = f2bf(s1[r]);
        }
        asm volatile("s_wait_dscnt 0" ::: "memory");   // same-wave DS RAW fence
        v16bf pf;
        {
            const u16* p = &pw[l16 * 32 + 16 * half];
            pf = mkfrag(*(const v8u*)p, *(const v8u*)(p + 8));
        }

        // ---- O += P @ V (8 d-chunks of 16) ----
#pragma unroll
        for (int j = 0; j < 8; ++j) {
            const u16* p = &Vt[(16 * j + l16) * LDV + 16 * half];
            o[j] = wmma_bf16(pf, mkfrag(*(const v8u*)p, *(const v8u*)(p + 8)), o[j]);
        }
        __syncthreads();
    }

    // ---- normalize + store bf16 ----
#pragma unroll
    for (int r = 0; r < 8; ++r) {
        const float inv = 1.0f / l[r];
        const int row = q0 + r + 8 * half;
        u16* dst = O + (baserow + row) * (size_t)A_D + headoff;
#pragma unroll
        for (int j = 0; j < 8; ++j)
            dst[16 * j + l16] = f2bf(o[j][r] * inv);
    }
}

// ---------------------------------------------------------------------------
// Launch: QKV projections (fp32->bf16), attention, output projection (->fp32)
// ---------------------------------------------------------------------------
extern "C" void kernel_launch(void* const* d_in, const int* in_sizes, int n_in,
                              void* d_out, int out_size, void* d_ws, size_t ws_size,
                              hipStream_t stream) {
    (void)in_sizes; (void)n_in; (void)out_size; (void)ws_size;
    const int B = 4, S = 2048, D = 2048;
    const int M = B * S;                 // 8192

    const float* q   = (const float*)d_in[0];
    const float* k   = (const float*)d_in[1];
    const float* v   = (const float*)d_in[2];
    const float* w_q = (const float*)d_in[3];
    const float* b_q = (const float*)d_in[4];
    const float* w_k = (const float*)d_in[5];
    const float* b_k = (const float*)d_in[6];
    const float* w_v = (const float*)d_in[7];
    const float* b_v = (const float*)d_in[8];
    const float* w_o = (const float*)d_in[9];
    const float* b_o = (const float*)d_in[10];

    const size_t ME = (size_t)M * D;     // elements per bf16 matrix
    u16* Qp  = (u16*)d_ws;
    u16* Kp  = Qp + ME;
    u16* Vp  = Kp + ME;
    u16* Abf = Vp + ME;

    dim3 gg(D / GBN, M / GBM);           // (16, 64)
    gemm_wmma<float, u16><<<gg, 256, 0, stream>>>(q, w_q, b_q, Qp, M, D, D);
    gemm_wmma<float, u16><<<gg, 256, 0, stream>>>(k, w_k, b_k, Kp, M, D, D);
    gemm_wmma<float, u16><<<gg, 256, 0, stream>>>(v, w_v, b_v, Vp, M, D, D);

    attn_wmma<<<dim3(64, S / 128), 256, 0, stream>>>(Qp, Kp, Vp, Abf);

    gemm_wmma<u16, float><<<gg, 256, 0, stream>>>(Abf, w_o, b_o, (float*)d_out, M, D, D);
}
